// ScaledDotProductAttention_19877108646161
// MI455X (gfx1250) — matrix operand order repaired and checked by reading, not compiled, not run on hardware
//
#include <hip/hip_runtime.h>
#include <hip/hip_bf16.h>
#include <math.h>

// CDNA5 / gfx1250: wave32, WMMA 16x16x32 bf16 with f32 accumulation.
typedef __bf16 v16bf __attribute__((ext_vector_type(16)));
typedef __bf16 v8bf  __attribute__((ext_vector_type(8)));
typedef float  v8f   __attribute__((ext_vector_type(8)));
typedef int    v4i   __attribute__((ext_vector_type(4)));

#define SS     2048
#define DK     64
#define NBATCH 16
#define ELEMS  (NBATCH * SS * DK)   // 2,097,152 elements of K / V / Q / out

static __device__ __forceinline__ v8f wmma_bf16(v16bf a, v16bf b, v8f c) {
  // 8 args: (neg_a, A, neg_b, B, c_mod, C, reuse_a, reuse_b)
  return __builtin_amdgcn_wmma_f32_16x16x32_bf16(false, a, false, b, (short)0, c,
                                                 false, false);
}

// ---------------- async global->LDS staging (ASYNCcnt path) ------------------

#if __has_builtin(__builtin_amdgcn_global_load_async_to_lds_b128)
#define K_ASYNC 1
#else
#define K_ASYNC 0
#endif

typedef __attribute__((address_space(1))) v4i gv4i_t;   // global int4
typedef __attribute__((address_space(3))) v4i lv4i_t;   // LDS int4

// Stage one 32x64 bf16 K tile (4KB): each of the 32 lanes owns one 128B row.
static __device__ __forceinline__ void stage_k(const __bf16* __restrict__ gk,
                                               __bf16* lk, int lane) {
#if K_ASYNC
  const char* g = (const char*)(gk + (size_t)lane * DK);
  char*       l = (char*)lk + lane * 128;
#pragma unroll
  for (int i = 0; i < 8; ++i) {
    __builtin_amdgcn_global_load_async_to_lds_b128(
        (gv4i_t*)(g + i * 16), (lv4i_t*)(l + i * 16), 0, 0);
  }
#else
  const uint4* g = (const uint4*)(gk + (size_t)lane * DK);
  uint4*       l = (uint4*)((char*)lk + lane * 128);
#pragma unroll
  for (int i = 0; i < 8; ++i) l[i] = g[i];
#endif
}

static __device__ __forceinline__ void wait_staged() {
#if K_ASYNC
#if __has_builtin(__builtin_amdgcn_s_wait_asynccnt)
  __builtin_amdgcn_s_wait_asynccnt(0);
#else
  asm volatile("s_wait_asynccnt 0x0" ::: "memory");
#endif
#else
  asm volatile("s_wait_dscnt 0x0" ::: "memory");
#endif
  __builtin_amdgcn_wave_barrier();
}

// ---------------- prep kernels: build bf16 operand copies in workspace -------

__global__ void prep_cvt_k(const float* __restrict__ K, __bf16* __restrict__ Kb) {
  int i = blockIdx.x * blockDim.x + threadIdx.x;
  if (i < ELEMS) Kb[i] = (__bf16)K[i];
}

// Vt[b][d][k] = V[b][k][d]  (per-batch transpose, bf16)
__global__ void prep_vt(const float* __restrict__ V, __bf16* __restrict__ Vt) {
  int i = blockIdx.x * blockDim.x + threadIdx.x;
  if (i < ELEMS) {
    int b   = i >> 17;          // / (2048*64)
    int rem = i & 131071;
    int k   = rem >> 6;
    int d   = rem & 63;
    Vt[((size_t)b << 17) + (size_t)d * SS + k] = (__bf16)V[i];
  }
}

// Rt[t][d] = Krelpos[d][t]  (transpose, bf16) -> contiguous B-operand rows
__global__ void prep_rt(const float* __restrict__ R, __bf16* __restrict__ Rt) {
  int i = blockIdx.x * blockDim.x + threadIdx.x;
  if (i < DK * SS) {
    int d = i >> 11;            // / 2048
    int t = i & 2047;
    Rt[(size_t)t * DK + d] = (__bf16)R[i];
  }
}

// ---------------- fused causal flash attention with relative-position bias ---

__global__ __launch_bounds__(128) void flash_relattn(
    const float* __restrict__ Q, const __bf16* __restrict__ Kb,
    const __bf16* __restrict__ Vt, const __bf16* __restrict__ Rt,
    float* __restrict__ Out)
{
  // Per-wave LDS: double-buffered K tile (2 x 4KB) + P round-trip slab (1KB)
  __shared__ __align__(16) __bf16 kstage[4][2][32 * DK];
  __shared__ __align__(16) __bf16 plds[4][16 * 32];

  const int wave = threadIdx.x >> 5;
  const int lane = threadIdx.x & 31;
  const int half = lane >> 4;     // which 16-lane half
  const int rr   = lane & 15;
  const int b    = blockIdx.y;                       // batch 0..15
  const int qw   = (blockIdx.x * 4 + wave) * 16;     // wave's query-row base

  // ---- Q tile (16x64) -> two bf16 A operands (16x32 each), A layout:
  // lane half h, row rr: elems 0..15 = K h*16.. (contiguous)
  v16bf aq[2];
  {
    const float* qrow = Q + ((size_t)(b * SS + qw + rr)) * DK;
#pragma unroll
    for (int c = 0; c < 2; ++c) {
      const float* p = qrow + c * 32 + half * 16;
#pragma unroll
      for (int e = 0; e < 8; ++e) {
        aq[c][e]     = (__bf16)p[e];
        aq[c][8 + e] = (__bf16)p[8 + e];
      }
    }
  }

  v8f o0 = {}, o1 = {}, o2 = {}, o3 = {};   // O accum, 16x64 f32 (C layout)
  float m[8], l[8];
#pragma unroll
  for (int v = 0; v < 8; ++v) { m[v] = -__builtin_inff(); l[v] = 0.0f; }

  const int nblk = (qw + 15) / 32 + 1;      // key blocks of 32 up to diagonal
  const __bf16* kbatch = Kb + (size_t)b * SS * DK;

  stage_k(kbatch, kstage[wave][0], lane);   // prefetch key block 0

  for (int blk = 0; blk < nblk; ++blk) {
    const int kn = blk * 32;

    wait_staged();                          // current K tile resident in LDS
    const __bf16* kt = kstage[wave][blk & 1];
    if (blk + 1 < nblk) {                   // async-stage next block
      stage_k(kbatch + (size_t)(kn + 32) * DK, kstage[wave][(blk + 1) & 1], lane);
      __builtin_prefetch(Vt + ((size_t)b << 17) + (size_t)rr * SS + kn + 32, 0, 0);
      __builtin_prefetch(Rt + (size_t)((SS - 1) - qw + kn + 32) * DK, 0, 0);
    }

    // ---- S = Q K^T : two 16x16 tiles (B operands from LDS K tile)
    v8f s0 = {}, s1 = {};
    {
      const __bf16* k0 = kt + (size_t)rr * DK + half * 16;
      const __bf16* k1 = k0 + 16 * DK;
      s0 = wmma_bf16(aq[0], *(const v16bf*)(k0),      s0);
      s0 = wmma_bf16(aq[1], *(const v16bf*)(k0 + 32), s0);
      s1 = wmma_bf16(aq[0], *(const v16bf*)(k1),      s1);
      s1 = wmma_bf16(aq[1], *(const v16bf*)(k1 + 32), s1);
    }

    // ---- relative-position bias: three shifted 16-col windows A/B/C
    //   window A: cols C0-15..C0,  B: C0..C0+15,  C: C0+16..C0+31
    {
      const int C0 = (SS - 1) - qw + kn;
      int colA = C0 - 15 + rr; colA = colA < 0 ? 0 : (colA > SS - 1 ? SS - 1 : colA);
      int colB = C0 + rr;      colB = colB < 0 ? 0 : (colB > SS - 1 ? SS - 1 : colB);
      int colC = C0 + 16 + rr; colC = colC < 0 ? 0 : (colC > SS - 1 ? SS - 1 : colC);
      const __bf16* ra = Rt + (size_t)colA * DK + half * 16;
      const __bf16* rb = Rt + (size_t)colB * DK + half * 16;
      const __bf16* rc = Rt + (size_t)colC * DK + half * 16;
      v8f pA = {}, pB = {}, pC = {};
      pA = wmma_bf16(aq[0], *(const v16bf*)(ra),      pA);
      pA = wmma_bf16(aq[1], *(const v16bf*)(ra + 32), pA);
      pB = wmma_bf16(aq[0], *(const v16bf*)(rb),      pB);
      pB = wmma_bf16(aq[1], *(const v16bf*)(rb + 32), pB);
      pC = wmma_bf16(aq[0], *(const v16bf*)(rc),      pC);
      pC = wmma_bf16(aq[1], *(const v16bf*)(rc + 32), pC);
#pragma unroll
      for (int v = 0; v < 8; ++v) {
        const int shift = rr - (v + 8 * half);          // jj - ii in [-15,15]
        int cA = shift + 15; cA = cA < 0 ? 0 : (cA > 15 ? 15 : cA);
        int cB = shift;      cB = cB < 0 ? 0 : (cB > 15 ? 15 : cB);
        int cD = shift + 16; cD = cD < 0 ? 0 : (cD > 15 ? 15 : cD);
        const float gA = __shfl(pA[v], cA, 16);         // tile0, shift < 0
        const float gB = __shfl(pB[v], cB, 16);         // tile0, shift >= 0
        const float gD = __shfl(pB[v], cD, 16);         // tile1, shift < 0
        const float gC = __shfl(pC[v], cB, 16);         // tile1, shift >= 0
        s0[v] += (shift >= 0) ? gB : gA;
        s1[v] += (shift >= 0) ? gC : gD;
      }
    }

    // ---- causal mask, 1/sqrt(dk) scale, online softmax
#pragma unroll
    for (int v = 0; v < 8; ++v) {
      const int rowg = qw + v + 8 * half;
      const int colg = kn + rr;
      s0[v] = (colg      <= rowg) ? s0[v] * 0.125f : -__builtin_inff();
      s1[v] = (colg + 16 <= rowg) ? s1[v] * 0.125f : -__builtin_inff();
      float t = fmaxf(s0[v], s1[v]);
#pragma unroll
      for (int off = 8; off; off >>= 1) t = fmaxf(t, __shfl_xor(t, off, 16));
      const float mn    = fmaxf(m[v], t);
      const float alpha = __expf(m[v] - mn);
      m[v]  = mn;
      s0[v] = __expf(s0[v] - mn);
      s1[v] = __expf(s1[v] - mn);
      float rs = s0[v] + s1[v];
#pragma unroll
      for (int off = 8; off; off >>= 1) rs += __shfl_xor(rs, off, 16);
      l[v] = l[v] * alpha + rs;
      o0[v] *= alpha; o1[v] *= alpha; o2[v] *= alpha; o3[v] *= alpha;
    }

    // ---- P (f32, C layout) -> bf16 A operand via per-wave LDS slab
    __bf16* myp = plds[wave];
#pragma unroll
    for (int v = 0; v < 8; ++v) {
      const int row = v + 8 * half;
      myp[row * 32 + rr]      = (__bf16)s0[v];
      myp[row * 32 + 16 + rr] = (__bf16)s1[v];
    }
    asm volatile("s_wait_dscnt 0x0" ::: "memory");  // per-wave LDS ordering
    __builtin_amdgcn_wave_barrier();
    v16bf ap;
    {
      const v8bf lo = *(const v8bf*)(myp + rr * 32 + half * 16);
      const v8bf hi = *(const v8bf*)(myp + rr * 32 + half * 16 + 8);
#pragma unroll
      for (int e = 0; e < 8; ++e) { ap[e] = lo[e]; ap[8 + e] = hi[e]; }
    }

    // ---- O += P * V   (V^T rows give contiguous B operands)
    {
      const __bf16* vb = Vt + ((size_t)b << 17) + (size_t)rr * SS + kn + half * 16;
      o0 = wmma_bf16(ap, *(const v16bf*)(vb),           o0);
      o1 = wmma_bf16(ap, *(const v16bf*)(vb + 16 * SS), o1);
      o2 = wmma_bf16(ap, *(const v16bf*)(vb + 32 * SS), o2);
      o3 = wmma_bf16(ap, *(const v16bf*)(vb + 48 * SS), o3);
    }
  }

  // ---- epilogue: normalize and store f32 output
#pragma unroll
  for (int v = 0; v < 8; ++v) {
    const float inv = 1.0f / l[v];
    float* orow = Out + ((size_t)(b * SS + qw + v + 8 * half)) * DK + rr;
    orow[0]  = o0[v] * inv;
    orow[16] = o1[v] * inv;
    orow[32] = o2[v] * inv;
    orow[48] = o3[v] * inv;
  }
}

// ---------------- launcher ---------------------------------------------------

extern "C" void kernel_launch(void* const* d_in, const int* in_sizes, int n_in,
                              void* d_out, int out_size, void* d_ws, size_t ws_size,
                              hipStream_t stream) {
  const float* Q = (const float*)d_in[0];
  const float* K = (const float*)d_in[1];
  const float* V = (const float*)d_in[2];
  const float* R = (const float*)d_in[3];   // Krelpos (64 x 2048)
  float* out = (float*)d_out;

  __bf16* Kb = (__bf16*)d_ws;               // 4 MB
  __bf16* Vt = Kb + ELEMS;                  // 4 MB
  __bf16* Rt = Vt + ELEMS;                  // 256 KB

  prep_cvt_k<<<ELEMS / 256, 256, 0, stream>>>(K, Kb);
  prep_vt   <<<ELEMS / 256, 256, 0, stream>>>(V, Vt);
  prep_rt   <<<(DK * SS) / 256, 256, 0, stream>>>(R, Rt);

  dim3 grid(SS / 64, NBATCH);               // 32 q-blocks x 16 batches
  flash_relattn<<<grid, 128, 0, stream>>>(Q, Kb, Vt, Rt, out);
}
